// RNN_77867757076551
// MI455X (gfx1250) — matrix operand order repaired and checked by reading, not compiled, not run on hardware
//
#include <hip/hip_runtime.h>

// ---------------------------------------------------------------------------
// Stacked Elman RNN (T=70, B=64, V=10000, E=512, H=1024, L=2) for gfx1250.
// f16 WMMA (v_wmma_f32_16x16x32_f16) for all GEMMs, f32 accumulate,
// f32 bias+tanh epilogue. Logits hoisted into one big batched GEMM with
// async global->LDS A staging (ASYNCcnt path) + register blocking.
// ---------------------------------------------------------------------------

typedef __attribute__((ext_vector_type(16))) _Float16 v16h;
typedef __attribute__((ext_vector_type(8)))  _Float16 v8h;
typedef __attribute__((ext_vector_type(8)))  float    v8f;

#define TDIM 70
#define BDIM 64
#define VDIM 10000
#define EDIM 512
#define HDIM 1024

// ---- WMMA fragment loaders (layouts per CDNA5 ISA §7.12.2) -----
// A: 16x32 f16, row-major source [M, ld]. lane(0..15)=M, lanes 16..31 same M
// but K+8/K+24 halves. Two 16B loads per lane.
__device__ __forceinline__ v16h load_a_frag(const _Float16* __restrict__ X,
                                            int ld, int mbase, int kbase,
                                            int lo, int hi) {
  union { v16h v; v8h h[2]; } u;
  const _Float16* p = X + (size_t)(mbase + lo) * ld + kbase + (hi << 3);
  u.h[0] = *(const v8h*)(p);        // K = kbase+hi*8   .. +7
  u.h[1] = *(const v8h*)(p + 16);   // K = kbase+hi*8+16 .. +23
  return u.v;
}

// B: 32x16 f16 needed as B[k][n] = W[n][k] with W row-major [N, ld] (i.e. the
// nn.Linear [out,in] weight used as x @ W^T). lane(0..15)=N, lanes 16..31 same N
// but K+8/K+24 halves (same as A). Two 16B loads per lane.
__device__ __forceinline__ v16h load_b_frag(const _Float16* __restrict__ W,
                                            int ld, int nbase, int kbase,
                                            int lo, int hi) {
  union { v16h v; v8h h[2]; } u;
  const _Float16* p = W + (size_t)(nbase + lo) * ld + kbase + (hi << 3);
  u.h[0] = *(const v8h*)(p);        // K = kbase+hi*8   .. +7
  u.h[1] = *(const v8h*)(p + 16);   // K = kbase+hi*8+16 .. +23
  return u.v;
}

// ---- small utility kernels -------------------------------------------------
__global__ void f32_to_f16_k(const float* __restrict__ src,
                             _Float16* __restrict__ dst, int n) {
  int i = blockIdx.x * blockDim.x + threadIdx.x;
  if (i < n) dst[i] = (_Float16)src[i];
}

__global__ void embed_gather_k(const int* __restrict__ idx,
                               const float* __restrict__ embW,
                               _Float16* __restrict__ dst) {
  size_t i = (size_t)blockIdx.x * blockDim.x + threadIdx.x;
  if (i >= (size_t)TDIM * BDIM * EDIM) return;
  size_t tb = i / EDIM;
  int e = (int)(i - tb * EDIM);
  dst[i] = (_Float16)embW[(size_t)idx[tb] * EDIM + e];
}

__global__ void copy_hidden_k(const float* __restrict__ h0,
                              const float* __restrict__ h1,
                              float* __restrict__ dst) {
  int i = blockIdx.x * blockDim.x + threadIdx.x;
  if (i < BDIM * HDIM) {
    dst[i] = h0[i];
    dst[BDIM * HDIM + i] = h1[i];
  }
}

// ---- one RNN layer for one timestep ---------------------------------------
// H[B,HDIM] = tanh(X[B,KX] @ Wx^T + Hprev[B,HDIM] @ Wh^T + bias)
// 256 output tiles (4 x 64); 4 waves/block, one 16x16 tile per wave, 64 blocks.
template <int KX>
__global__ void __launch_bounds__(128)
rnn_layer_wmma(const _Float16* __restrict__ Xh,
               const _Float16* __restrict__ Hph,
               const _Float16* __restrict__ Wxh,   // [HDIM, KX]
               const _Float16* __restrict__ Whh,   // [HDIM, HDIM]
               const float* __restrict__ bias,     // [HDIM]
               float* __restrict__ Hf32,           // [B, HDIM]
               _Float16* __restrict__ Hf16) {      // [B, HDIM]
  const int lane = threadIdx.x & 31;
  const int wave = threadIdx.x >> 5;
  const int tile = blockIdx.x * 4 + wave;          // 0..255
  const int mbase = (tile & 3) << 4;               // B tile
  const int nbase = (tile >> 2) << 4;              // H tile
  const int hi = lane >> 4;
  const int lo = lane & 15;

  v8f acc = {};
#pragma unroll 4
  for (int k = 0; k < KX; k += 32) {
    v16h a = load_a_frag(Xh, KX, mbase, k, lo, hi);
    v16h b = load_b_frag(Wxh, KX, nbase, k, lo, hi);
    acc = __builtin_amdgcn_wmma_f32_16x16x32_f16(false, a, false, b,
                                                 (short)0, acc, false, false);
  }
#pragma unroll 4
  for (int k = 0; k < HDIM; k += 32) {
    v16h a = load_a_frag(Hph, HDIM, mbase, k, lo, hi);
    v16h b = load_b_frag(Whh, HDIM, nbase, k, lo, hi);
    acc = __builtin_amdgcn_wmma_f32_16x16x32_f16(false, a, false, b,
                                                 (short)0, acc, false, false);
  }

  const int col = nbase + lo;
  const float bn = bias[col];
#pragma unroll
  for (int r = 0; r < 8; ++r) {
    const int row = mbase + (hi << 3) + r;     // C layout: VGPR r -> M=r / r+8
    float v = tanhf(acc[r] + bn);
    Hf32[(size_t)row * HDIM + col] = v;
    Hf16[(size_t)row * HDIM + col] = (_Float16)v;
  }
}

// ---- batched logits GEMM:  C[T*B, V] = H1all @ Wy^T + by -------------------
// Block = 8 waves; one 16-row A tile staged to LDS via async copy, shared by
// the whole block. Each wave owns 5 accumulators (16x80 of C), reusing the
// LDS A fragment 5x per k-step. V = 10000 = 125 groups * 80 cols; 16 blocks
// of 8 waves cover the 125 groups (last 3 waves idle, wave-uniform).
#define NGROUPS 125
#define WAVE_NT 5

__global__ void __launch_bounds__(256)
logits_wmma(const _Float16* __restrict__ A,      // [T*B, HDIM]
            const _Float16* __restrict__ W,      // [VDIM, HDIM]
            const float* __restrict__ bias,      // [VDIM]
            float* __restrict__ C) {             // [T*B, VDIM]
  __shared__ _Float16 As[16 * HDIM];             // 32 KB A tile (of 320 KB WGP LDS)

  const int tid   = threadIdx.x;
  const int lane  = tid & 31;
  const int wave  = tid >> 5;
  const int mtile = blockIdx.x >> 4;             // 280 M tiles
  const int gblk  = blockIdx.x & 15;             // 16 N-groups-of-8-waves
  const int mbase = mtile << 4;

  // ---- cooperative async copy: 16 x 1024 f16 A rows (contiguous) -> LDS ----
  {
    const _Float16* src = A + (size_t)mbase * HDIM;
    const unsigned  lds0 = (unsigned)(uintptr_t)As;
#pragma unroll
    for (int j = 0; j < 8; ++j) {
      const int e = (j * 256 + tid) * 8;         // halfs; 8*256*8 = 16384 = tile
      const unsigned dst = lds0 + e * 2;
      const _Float16* g = src + e;
      asm volatile("global_load_async_to_lds_b128 %0, %1, off"
                   :: "v"(dst), "v"(g) : "memory");
    }
    asm volatile("s_wait_asynccnt 0x0" ::: "memory");
    __syncthreads();
  }

  const int ngroup = gblk * 8 + wave;
  if (ngroup < NGROUPS) {
    const int hi = lane >> 4;
    const int lo = lane & 15;
    const int nbase = ngroup * (WAVE_NT * 16);

    v8f acc[WAVE_NT];
#pragma unroll
    for (int t = 0; t < WAVE_NT; ++t) {
      v8f z = {};
      acc[t] = z;
    }

    for (int k = 0; k < HDIM; k += 32) {
      // A fragment from LDS (ds_load_b128 x2), reused across all 5 N tiles.
      union { v16h v; v8h h[2]; } ua;
      const _Float16* ap = As + lo * HDIM + k + (hi << 3);
      ua.h[0] = *(const v8h*)(ap);
      ua.h[1] = *(const v8h*)(ap + 16);
#pragma unroll
      for (int t = 0; t < WAVE_NT; ++t) {
        __builtin_prefetch(W + (size_t)(nbase + (t << 4) + lo) * HDIM + k + 512,
                           0, 3);                // near prefetch of next B rows
        v16h b = load_b_frag(W, HDIM, nbase + (t << 4), k, lo, hi);
        acc[t] = __builtin_amdgcn_wmma_f32_16x16x32_f16(
            false, ua.v, false, b, (short)0, acc[t], false, false);
      }
    }

#pragma unroll
    for (int t = 0; t < WAVE_NT; ++t) {
      const int col = nbase + (t << 4) + lo;
      const float bn = bias[col];
#pragma unroll
      for (int r = 0; r < 8; ++r) {
        const int row = mbase + (hi << 3) + r;
        C[(size_t)row * VDIM + col] = acc[t][r] + bn;
      }
    }
  }
}

// ---------------------------------------------------------------------------
extern "C" void kernel_launch(void* const* d_in, const int* in_sizes, int n_in,
                              void* d_out, int out_size, void* d_ws,
                              size_t ws_size, hipStream_t stream) {
  (void)in_sizes; (void)n_in; (void)out_size; (void)ws_size;
  const int*   inputs = (const int*)d_in[0];
  const float* hidden = (const float*)d_in[1];
  const float* embW   = (const float*)d_in[2];
  const float* Wx0    = (const float*)d_in[3];
  const float* Wx1    = (const float*)d_in[4];
  const float* Wh0    = (const float*)d_in[5];
  const float* bh0    = (const float*)d_in[6];
  const float* Wh1    = (const float*)d_in[7];
  const float* bh1    = (const float*)d_in[8];
  const float* Wy     = (const float*)d_in[9];
  const float* by     = (const float*)d_in[10];
  float* out = (float*)d_out;

  // ---- carve workspace (~42 MB) -------------------------------------------
  char* ws = (char*)d_ws;
  size_t off = 0;
  auto carve16 = [&](size_t elems) -> _Float16* {
    off = (off + 255) & ~(size_t)255;
    _Float16* p = (_Float16*)(ws + off);
    off += elems * sizeof(_Float16);
    return p;
  };
  auto carve32 = [&](size_t elems) -> float* {
    off = (off + 255) & ~(size_t)255;
    float* p = (float*)(ws + off);
    off += elems * sizeof(float);
    return p;
  };
  _Float16* Wx0h  = carve16((size_t)HDIM * EDIM);
  _Float16* Wh0h  = carve16((size_t)HDIM * HDIM);
  _Float16* Wx1h  = carve16((size_t)HDIM * HDIM);
  _Float16* Wh1h  = carve16((size_t)HDIM * HDIM);
  _Float16* Wyh   = carve16((size_t)VDIM * HDIM);
  _Float16* embh  = carve16((size_t)TDIM * BDIM * EDIM);
  _Float16* H1all = carve16((size_t)TDIM * BDIM * HDIM);  // h1 log = layer1 history
  _Float16* h0h[2] = { carve16((size_t)BDIM * HDIM), carve16((size_t)BDIM * HDIM) };
  _Float16* h1h0  = carve16((size_t)BDIM * HDIM);         // initial h1 only
  float* h0f32 = carve32((size_t)BDIM * HDIM);
  float* h1f32 = carve32((size_t)BDIM * HDIM);

  // ---- weight / init conversions ------------------------------------------
  auto cvt = [&](const float* s, _Float16* d, size_t n) {
    f32_to_f16_k<<<(int)((n + 255) / 256), 256, 0, stream>>>(s, d, (int)n);
  };
  cvt(Wx0, Wx0h, (size_t)HDIM * EDIM);
  cvt(Wh0, Wh0h, (size_t)HDIM * HDIM);
  cvt(Wx1, Wx1h, (size_t)HDIM * HDIM);
  cvt(Wh1, Wh1h, (size_t)HDIM * HDIM);
  cvt(Wy,  Wyh,  (size_t)VDIM * HDIM);
  cvt(hidden,                         h0h[0], (size_t)BDIM * HDIM);
  cvt(hidden + (size_t)BDIM * HDIM,   h1h0,   (size_t)BDIM * HDIM);

  // ---- embedding gather ----------------------------------------------------
  {
    size_t n = (size_t)TDIM * BDIM * EDIM;
    embed_gather_k<<<(int)((n + 255) / 256), 256, 0, stream>>>(inputs, embW, embh);
  }

  // ---- sequential recurrence (70 x 2 small WMMA GEMMs) ---------------------
  for (int t = 0; t < TDIM; ++t) {
    const _Float16* xt = embh + (size_t)t * BDIM * EDIM;
    rnn_layer_wmma<EDIM><<<64, 128, 0, stream>>>(
        xt, h0h[t & 1], Wx0h, Wh0h, bh0, h0f32, h0h[(t + 1) & 1]);
    const _Float16* h1prev =
        (t == 0) ? h1h0 : (H1all + (size_t)(t - 1) * BDIM * HDIM);
    rnn_layer_wmma<HDIM><<<64, 128, 0, stream>>>(
        h0h[(t + 1) & 1], h1prev, Wx1h, Wh1h, bh1, h1f32,
        H1all + (size_t)t * BDIM * HDIM);
  }

  // ---- all logits in one compute-bound WMMA GEMM ---------------------------
  // 280 M tiles x 16 wavegroup-blocks = 4480 blocks of 8 waves.
  logits_wmma<<<280 * 16, 256, 0, stream>>>(H1all, Wyh, by, out);

  // ---- final hidden state -> tail of d_out ---------------------------------
  copy_hidden_k<<<(BDIM * HDIM + 255) / 256, 256, 0, stream>>>(
      h0f32, h1f32, out + (size_t)TDIM * BDIM * VDIM);
}
